// LFDv2_9586367005084
// MI455X (gfx1250) — 8-bit matrix operand order repaired and checked by reading, compiled, not run on hardware
//
#include <hip/hip_runtime.h>
#include <hip/hip_bf16.h>

typedef __attribute__((ext_vector_type(8))) int v8i;

#define NCLS 80
#define NGT  64
#define PTS_PER_WAVE  16
#define WAVES_PER_BLK 8
#define PTS_PER_BLK   (PTS_PER_WAVE * WAVES_PER_BLK)

__global__ __launch_bounds__(256) void lfd_assign_wmma_kernel(
    const float* __restrict__ points,
    const float* __restrict__ reg_ranges,
    const float* __restrict__ gray_ranges,
    const float* __restrict__ strides,
    const float* __restrict__ gt_bboxes,
    const int*   __restrict__ gt_labels,
    float* __restrict__ cls_out,
    float* __restrict__ reg_out,
    int P)
{
    __shared__ float4 sBox[NGT];
    __shared__ float4 sCore[NGT];
    __shared__ int    sLbl[NGT];
    __shared__ uint32_t sGreen[2], sGray[2];
    __shared__ float2 sRes[PTS_PER_BLK];
    __shared__ uint32_t sB[5 * 32 * 8];

    const int b      = blockIdx.y;
    const int tid    = threadIdx.x;
    const int pbase0 = blockIdx.x * PTS_PER_BLK;

    const float rlo = reg_ranges[2*pbase0],  rhi = reg_ranges[2*pbase0+1];
    const float glo = gray_ranges[2*pbase0], ghi = gray_ranges[2*pbase0+1];
    const float s2u = 0.5f * strides[pbase0];

    if (tid < NGT) {
        const float4 bx = ((const float4*)gt_bboxes)[(size_t)b * NGT + tid];
        float4 box, core;
        box.x = bx.x;
        box.y = bx.y;
        box.z = bx.x + bx.z - 1.0f;
        box.w = bx.y + bx.w - 1.0f;
        const float cx = bx.x + 0.5f * bx.z;
        const float cy = bx.y + 0.5f * bx.w;
        core.x = cx - s2u;  core.y = cx + s2u;
        core.z = cy - s2u;  core.w = cy + s2u;
        sBox[tid]  = box;
        sCore[tid] = core;
        sLbl[tid]  = gt_labels[(size_t)b * NGT + tid];

        const float meas = fmaxf(bx.z, bx.w);
        const bool grn = (rlo <= meas) && (meas <= rhi);
        const bool gry = ((glo <= meas) && (meas <  rlo)) ||
                         ((rhi <  meas) && (meas <= ghi));
        const uint32_t bg = __builtin_amdgcn_ballot_w32(grn);
        const uint32_t by = __builtin_amdgcn_ballot_w32(gry);
        if ((tid & 31) == 0) { sGreen[tid >> 5] = bg; sGray[tid >> 5] = by; }
    }
    __syncthreads();

    #pragma unroll
    for (int i = 0; i < 5; ++i) {
        const int idx = i * 256 + tid;
        const int cg  = idx >> 8;
        const int ln  = (idx >> 3) & 31;
        const int v   = idx & 7;
        const int hib = ln >> 4;
        const int c   = cg * 16 + (ln & 15);
        const int kb  = 16*(v >> 1) + 4*(v & 1) + 8*hib;
        uint32_t w = 0;
        #pragma unroll
        for (int t = 0; t < 4; ++t)
            w |= ((sLbl[kb + t] == c) ? 1u : 0u) << (8*t);
        sB[idx] = w;
    }

    const int  lane  = tid & 31;
    const int  wv    = tid >> 5;
    const int  hi    = lane >> 4;
    const int  m     = lane & 15;
    const int  pbase = pbase0 + wv * PTS_PER_WAVE;
    const bool valid = pbase < P;
    const int  p     = pbase + m;
    const int  pc    = (p < P) ? p : (P - 1);

    const float px = points[2*pc], py = points[2*pc+1];

    __syncthreads();

    const int sh = hi * 8;
    const uint32_t gmv0 = sGreen[0], gmv1 = sGreen[1];
    const uint32_t ymv0 = sGray[0],  ymv1 = sGray[1];
    const uint32_t gm0 = gmv0 >> sh, gm1 = gmv1 >> sh;
    const uint32_t ym0 = ymv0 >> sh, ym1 = ymv1 >> sh;
    const uint32_t act0 = (uint32_t)__builtin_amdgcn_readfirstlane((int)(gmv0 | ymv0));
    const uint32_t act1 = (uint32_t)__builtin_amdgcn_readfirstlane((int)(gmv1 | ymv1));

    float    best  = 0.0f;
    int      bestg = 0;
    uint32_t av[8];

    #pragma unroll
    for (int blk = 0; blk < 4; ++blk) {
      const uint32_t gm  = (blk < 2) ? gm0  : gm1;
      const uint32_t ym  = (blk < 2) ? ym0  : ym1;
      const uint32_t act = (blk < 2) ? act0 : act1;
      #pragma unroll
      for (int hh = 0; hh < 2; ++hh) {
        const int wbase = (blk & 1)*16 + hh*4;
        uint32_t ab = 0;
        if (((act >> wbase) & 0x0F0Fu) != 0u) {
          #pragma unroll
          for (int t = 0; t < 4; ++t) {
            const int g    = blk*16 + hi*8 + hh*4 + t;
            const int bitc = wbase + t;
            const float4 bxv = sBox[g];
            const float4 cov = sCore[g];
            const float d1 = px - bxv.x;
            const float d2 = py - bxv.y;
            const float d3 = bxv.z - px;
            const float d4 = bxv.w - py;

            const float lrmin = fminf(d1, d3), lrmax = fmaxf(d1, d3);
            const float tbmin = fminf(d2, d4), tbmax = fmaxf(d2, d4);
            const float mn  = fminf(lrmin, tbmin);
            const bool  hit = (mn >= 0.0f);

            const float num = lrmin * tbmin;
            const float den = fmaxf(lrmax, 0.01f) * fmaxf(tbmax, 0.01f);
            const float sr  = __builtin_amdgcn_sqrtf(num * __builtin_amdgcn_rcpf(den));

            const bool core = (px >= cov.x) && (px <= cov.y) &&
                              (py >= cov.z) && (py <= cov.w);
            const bool hg = hit && (((gm >> bitc) & 1u) != 0u);
            const float sc = hg ? (core ? 1.0f : sr) : 0.0f;

            if (sc > best) { best = sc; bestg = g; }
            const uint32_t gb = ((ym >> bitc) & 1u) << (8*t);
            ab |= hit ? gb : 0u;
          }
        }
        av[2*blk + hh] = ab;
      }
    }

    const float ob = __shfl_xor(best, 16, 32);
    const int   og = __shfl_xor(bestg, 16, 32);
    if (ob > best || (ob == best && og < bestg)) { best = ob; bestg = og; }

    if (hi == 0) {
        float2 res;
        res.x = best;
        res.y = __int_as_float(sLbl[bestg]);
        sRes[wv*PTS_PER_WAVE + m] = res;
        if (valid) {
            const bool pos = best > 0.0f;
            const float4 bw = sBox[bestg];
            float4 rg;
            rg.x = pos ? (px - bw.x) : 0.0f;
            rg.y = pos ? (py - bw.y) : 0.0f;
            rg.z = pos ? (bw.z - px) : 0.0f;
            rg.w = pos ? (bw.w - py) : 0.0f;
            ((float4*)reg_out)[(size_t)b * P + p] = rg;
        }
    }
    __syncthreads();

    v8i A;
    #pragma unroll
    for (int v = 0; v < 8; ++v) A[v] = (int)av[v];

    v8i D[5];
    #pragma unroll
    for (int cg = 0; cg < 5; ++cg) {
        const uint32_t* bp = &sB[(cg*32 + lane) * 8];
        const int4 b0 = ((const int4*)bp)[0];
        const int4 b1 = ((const int4*)bp)[1];
        v8i Bv;
        Bv[0] = b0.x; Bv[1] = b0.y; Bv[2] = b0.z; Bv[3] = b0.w;
        Bv[4] = b1.x; Bv[5] = b1.y; Bv[6] = b1.z; Bv[7] = b1.w;
        v8i Cz = {};
        D[cg] = __builtin_amdgcn_wmma_i32_16x16x64_iu8(
                    false, A, false, Bv, Cz, false, false);
    }

    if (valid) {
        float* __restrict__ crow =
            cls_out + ((size_t)b * P + pbase + 8*hi) * NCLS + m;
        const float2* __restrict__ rres = &sRes[wv*PTS_PER_WAVE + 8*hi];
        #pragma unroll
        for (int r = 0; r < 8; ++r) {
            const float2 res = rres[r];
            const float  scv = res.x;
            const int    mc  = (scv > 0.0f) ? __float_as_int(res.y) : -1;
            #pragma unroll
            for (int cg = 0; cg < 5; ++cg) {
                const int   c   = cg*16 + m;
                const float val = (mc == c) ? scv
                                            : ((D[cg][r] > 0) ? -1.0f : 0.0f);
                crow[r*NCLS + cg*16] = val;
            }
        }
    }
}

extern "C" void kernel_launch(void* const* d_in, const int* in_sizes, int n_in,
                              void* d_out, int out_size, void* d_ws, size_t ws_size,
                              hipStream_t stream) {
    (void)n_in; (void)out_size; (void)d_ws; (void)ws_size;
    const float* points = (const float*)d_in[0];
    const float* rr     = (const float*)d_in[1];
    const float* gr     = (const float*)d_in[2];
    const float* st     = (const float*)d_in[3];
    const float* gtb    = (const float*)d_in[4];
    const int*   gtl    = (const int*)d_in[5];

    const int P = in_sizes[3];
    const int B = in_sizes[5] / NGT;

    float* cls = (float*)d_out;
    float* reg = cls + (size_t)B * P * NCLS;

    dim3 grid((P + PTS_PER_BLK - 1) / PTS_PER_BLK, B);
    lfd_assign_wmma_kernel<<<grid, 256, 0, stream>>>(
        points, rr, gr, st, gtb, gtl, cls, reg, P);
}
